// GCNWithTOGL_69174743269593
// MI455X (gfx1250) — matrix operand order repaired and checked by reading, not compiled, not run on hardware
//
#include <hip/hip_runtime.h>

// ---------------- static problem sizes (match reference) ----------------
#define HID     128
#define NFILT   16
#define NNODES  131072          // 1024 graphs * 128 nodes
#define NEDGES  1048576         // NNODES * 8
#define NGRAPH  1024
#define OUTCH   10

typedef __attribute__((ext_vector_type(16))) __bf16 v16bf;
typedef __attribute__((ext_vector_type(8)))  float  v8f;

// monotonic float <-> int key (for atomicMax on signed floats)
__device__ __forceinline__ int fkey(float x) {
  unsigned u = __float_as_uint(x);
  return (int)((u & 0x80000000u) ? ~u : (u | 0x80000000u));
}
__device__ __forceinline__ float funkey(int k) {
  unsigned u = (unsigned)k;
  u = (u & 0x80000000u) ? (u ^ 0x80000000u) : ~u;
  return __uint_as_float(u);
}

// ---------------- weight packing into WMMA B-fragment layout ----------------
// B (32x16 bf16 per k-tile): lane holds column n = lane&15;
// lane group (lane>>4) selects K half; element i <-> k = ktile*32 + (lane>>4)*16 + i.
__global__ void pack_b_kernel(const float* __restrict__ W, __bf16* __restrict__ Bhi,
                              __bf16* __restrict__ Blo, int KT, int NT, int Ncols) {
  int t = blockIdx.x * blockDim.x + threadIdx.x;
  int total = NT * KT * 32;
  if (t >= total) return;
  int lane = t & 31;
  int kt   = (t >> 5) % KT;
  int nt   = (t >> 5) / KT;
  int n     = nt * 16 + (lane & 15);
  int kbase = kt * 32 + (lane >> 4) * 16;
  size_t o = ((size_t)(nt * KT + kt) * 32 + lane) * 16;
#pragma unroll
  for (int i = 0; i < 16; ++i) {
    float w = W[(size_t)(kbase + i) * Ncols + n];
    __bf16 h = (__bf16)w;
    Bhi[o + i] = h;
    Blo[o + i] = (__bf16)(w - (float)h);
  }
}

// ---------------- WMMA GEMM: out[N x NT*16] = A[N x KT*32] @ B (+bias) ----------------
// 4 waves / block, each wave owns 16 rows and all NT column tiles.
// bf16 hi/lo split: acc += Ahi*Bhi + Alo*Bhi + Ahi*Blo  (fp32 accumulate, ~fp32 accuracy)
template<int NT>
__global__ void __launch_bounds__(128)
wmma_gemm_kernel(const float* __restrict__ A, const __bf16* __restrict__ Bhi,
                 const __bf16* __restrict__ Blo, const float* __restrict__ bias,
                 float* __restrict__ out, int KT) {
  const int K     = KT * 32;
  const int Ncols = NT * 16;
  int lane = threadIdx.x & 31;
  int wave = threadIdx.x >> 5;
  int rowBase = (blockIdx.x * 4 + wave) * 16;
  int mrow = lane & 15;
  int grp  = lane >> 4;
  const float* Arow = A + (size_t)(rowBase + mrow) * K;

  v8f acc[NT] = {};

  for (int kt = 0; kt < KT; ++kt) {
    // A fragment (16x32 bf16): lane = row; elem i <-> k = kt*32 + grp*16 + i
    const float* ap = Arow + kt * 32 + grp * 16;
    float4 q0 = *(const float4*)(ap);
    float4 q1 = *(const float4*)(ap + 4);
    float4 q2 = *(const float4*)(ap + 8);
    float4 q3 = *(const float4*)(ap + 12);
    float av[16] = {q0.x,q0.y,q0.z,q0.w, q1.x,q1.y,q1.z,q1.w,
                    q2.x,q2.y,q2.z,q2.w, q3.x,q3.y,q3.z,q3.w};
    v16bf ahi, alo;
#pragma unroll
    for (int i = 0; i < 16; ++i) {
      __bf16 hv = (__bf16)av[i];
      ahi[i] = hv;
      alo[i] = (__bf16)(av[i] - (float)hv);
    }
#pragma unroll
    for (int nt = 0; nt < NT; ++nt) {
      size_t bo = ((size_t)(nt * KT + kt) * 32 + lane) * 16;
      v16bf bhi = *(const v16bf*)(Bhi + bo);
      v16bf blo = *(const v16bf*)(Blo + bo);
      acc[nt] = __builtin_amdgcn_wmma_f32_16x16x32_bf16(false, ahi, false, bhi,
                                                        (short)0, acc[nt], false, false);
      acc[nt] = __builtin_amdgcn_wmma_f32_16x16x32_bf16(false, alo, false, bhi,
                                                        (short)0, acc[nt], false, false);
      acc[nt] = __builtin_amdgcn_wmma_f32_16x16x32_bf16(false, ahi, false, blo,
                                                        (short)0, acc[nt], false, false);
    }
  }
  // D layout: lane = col (lane&15); VGPR v = row v + grp*8
#pragma unroll
  for (int nt = 0; nt < NT; ++nt) {
    int col = nt * 16 + mrow;
    float bv = (bias != nullptr) ? bias[col] : 0.0f;
#pragma unroll
    for (int v = 0; v < 8; ++v) {
      int r = rowBase + grp * 8 + v;
      out[(size_t)r * Ncols + col] = acc[nt][v] + bv;
    }
  }
}

// ---------------- embedding: h = emb_classes[x0] + emb_neigh[x1] ----------------
__global__ void embed_kernel(const int* __restrict__ x, const float* __restrict__ ec,
                             const float* __restrict__ en, float* __restrict__ h) {
  int t = blockIdx.x * blockDim.x + threadIdx.x;
  if (t >= NNODES * 32) return;
  int node = t >> 5, c4 = (t & 31) * 4;
  int cls = x[node * 2 + 0];
  int ng  = x[node * 2 + 1];
  float4 a = *(const float4*)(ec + (size_t)cls * HID + c4);
  float4 b = *(const float4*)(en + (size_t)ng  * HID + c4);
  float4 r; r.x = a.x + b.x; r.y = a.y + b.y; r.z = a.z + b.z; r.w = a.w + b.w;
  *(float4*)(h + (size_t)node * HID + c4) = r;
}

// ---------------- degree / dinv ----------------
__global__ void fill1_kernel(float* __restrict__ p, int n) {
  int t = blockIdx.x * blockDim.x + threadIdx.x;
  if (t < n) p[t] = 1.0f;
}
__global__ void deg_count_kernel(const int* __restrict__ ei, float* __restrict__ deg) {
  int e = blockIdx.x * blockDim.x + threadIdx.x;
  if (e < NEDGES) atomicAdd(deg + ei[NEDGES + e], 1.0f);
}
__global__ void rsqrt_kernel(float* __restrict__ p, int n) {
  int t = blockIdx.x * blockDim.x + threadIdx.x;
  if (t < n) p[t] = rsqrtf(p[t]);
}

// ---------------- GCN aggregation ----------------
__global__ void selfterm_kernel(const float* __restrict__ m, const float* __restrict__ dinv,
                                float* __restrict__ agg) {
  int t = blockIdx.x * blockDim.x + threadIdx.x;
  if (t >= NNODES * 32) return;
  int node = t >> 5, c4 = (t & 31) * 4;
  float w = dinv[node]; w *= w;
  float4 v = *(const float4*)(m + (size_t)node * HID + c4);
  v.x *= w; v.y *= w; v.z *= w; v.w *= w;
  *(float4*)(agg + (size_t)node * HID + c4) = v;
}

__global__ void scatter_kernel(const int* __restrict__ ei, const float* __restrict__ m,
                               const float* __restrict__ dinv, float* __restrict__ agg) {
  int t = blockIdx.x * blockDim.x + threadIdx.x;
  int e = t >> 5;
  if (e >= NEDGES) return;
  int lane = t & 31;
  int s = ei[e], d = ei[NEDGES + e];
  float w = dinv[s] * dinv[d];
  float4 v = *(const float4*)(m + (size_t)s * HID + lane * 4);
  float* o = agg + (size_t)d * HID + lane * 4;
  atomicAdd(o + 0, v.x * w);
  atomicAdd(o + 1, v.y * w);
  atomicAdd(o + 2, v.z * w);
  atomicAdd(o + 3, v.w * w);
}

// ---------------- fused bias + ReLU + residual + LayerNorm (wave per node) ----------------
__global__ void ln_kernel(const float* __restrict__ agg, const float* __restrict__ bias,
                          const float* __restrict__ gamma, const float* __restrict__ beta,
                          float* __restrict__ h) {
  int t = blockIdx.x * blockDim.x + threadIdx.x;
  int node = t >> 5;
  if (node >= NNODES) return;
  int c4 = (t & 31) * 4;
  float4 a  = *(const float4*)(agg + (size_t)node * HID + c4);
  float4 hv = *(const float4*)(h   + (size_t)node * HID + c4);
  float4 bb = *(const float4*)(bias + c4);
  float v0 = hv.x + fmaxf(a.x + bb.x, 0.0f);
  float v1 = hv.y + fmaxf(a.y + bb.y, 0.0f);
  float v2 = hv.z + fmaxf(a.z + bb.z, 0.0f);
  float v3 = hv.w + fmaxf(a.w + bb.w, 0.0f);
  float s  = v0 + v1 + v2 + v3;
  float ss = v0*v0 + v1*v1 + v2*v2 + v3*v3;
#pragma unroll
  for (int o = 16; o >= 1; o >>= 1) {
    s  += __shfl_xor(s,  o, 32);
    ss += __shfl_xor(ss, o, 32);
  }
  float mu  = s * (1.0f / HID);
  float var = ss * (1.0f / HID) - mu * mu;
  float rs  = rsqrtf(var + 1e-5f);
  float4 g  = *(const float4*)(gamma + c4);
  float4 be = *(const float4*)(beta  + c4);
  float4 r;
  r.x = (v0 - mu) * rs * g.x + be.x;
  r.y = (v1 - mu) * rs * g.y + be.y;
  r.z = (v2 - mu) * rs * g.z + be.z;
  r.w = (v3 - mu) * rs * g.w + be.w;
  *(float4*)(h + (size_t)node * HID + c4) = r;
}

// ---------------- TOGL persistence ----------------
__global__ void dkey_init_kernel(const float* __restrict__ f, int* __restrict__ dk) {
  int i = blockIdx.x * blockDim.x + threadIdx.x;
  if (i < NNODES * NFILT) dk[i] = fkey(f[i]);
}
__global__ void edge_max_kernel(const int* __restrict__ ei, const float* __restrict__ f,
                                int* __restrict__ dk) {
  int t = blockIdx.x * blockDim.x + threadIdx.x;
  if (t >= NEDGES * NFILT) return;
  int e = t >> 4, j = t & 15;
  int s = ei[e], d = ei[NEDGES + e];
  float em = fmaxf(f[(size_t)s * NFILT + j], f[(size_t)d * NFILT + j]);
  int k = fkey(em);
  atomicMax(dk + (size_t)d * NFILT + j, k);
  atomicMax(dk + (size_t)s * NFILT + j, k);
}
__global__ void pairs_kernel(const float* __restrict__ f, const int* __restrict__ dk,
                             float* __restrict__ prs) {
  int t = blockIdx.x * blockDim.x + threadIdx.x;
  if (t >= NNODES * NFILT) return;
  int n = t >> 4, j = t & 15;
  prs[(size_t)n * 32 + j]      = f[(size_t)n * NFILT + j];
  prs[(size_t)n * 32 + 16 + j] = funkey(dk[(size_t)n * NFILT + j]);
}
__global__ void add_relu_kernel(const float* __restrict__ t4, float* __restrict__ h4) {
  int i = blockIdx.x * blockDim.x + threadIdx.x;
  if (i >= NNODES * HID / 4) return;
  float4 a = ((const float4*)t4)[i];
  float4 b = ((float4*)h4)[i];
  b.x += fmaxf(a.x, 0.0f);
  b.y += fmaxf(a.y, 0.0f);
  b.z += fmaxf(a.z, 0.0f);
  b.w += fmaxf(a.w, 0.0f);
  ((float4*)h4)[i] = b;
}

// ---------------- readout: roots @ w_out ----------------
__global__ void out_kernel(const float* __restrict__ h, const float* __restrict__ w,
                           float* __restrict__ out) {
  int t = blockIdx.x * blockDim.x + threadIdx.x;
  if (t >= NGRAPH * OUTCH) return;
  int g = t / OUTCH, o = t % OUTCH;
  const float* row = h + (size_t)g * 128 * HID;  // node 0 of graph g
  float s = 0.0f;
#pragma unroll 4
  for (int c = 0; c < HID; ++c) s += row[c] * w[c * OUTCH + o];
  out[t] = s;
}

// ---------------- launcher ----------------
extern "C" void kernel_launch(void* const* d_in, const int* in_sizes, int n_in,
                              void* d_out, int out_size, void* d_ws, size_t ws_size,
                              hipStream_t stream) {
  (void)in_sizes; (void)n_in; (void)out_size; (void)ws_size;
  const int*   x       = (const int*)  d_in[0];
  const int*   ei      = (const int*)  d_in[1];
  // d_in[2] = batch (unused; implicit in node ordering)
  const float* emb_c   = (const float*)d_in[3];
  const float* emb_n   = (const float*)d_in[4];
  const float* gcn_w   = (const float*)d_in[5];
  const float* gcn_b   = (const float*)d_in[6];
  const float* ln_g    = (const float*)d_in[7];
  const float* ln_b    = (const float*)d_in[8];
  const float* togl_wf = (const float*)d_in[9];
  const float* togl_bf = (const float*)d_in[10];
  const float* togl_wp = (const float*)d_in[11];
  const float* togl_bp = (const float*)d_in[12];
  const float* w_out   = (const float*)d_in[13];
  float* out = (float*)d_out;

  // workspace bump allocator (256B aligned)
  char* wsp = (char*)d_ws;
  auto walloc = [&](size_t bytes) -> void* {
    void* p = (void*)wsp;
    wsp += (bytes + 255) & ~(size_t)255;
    return p;
  };
  float* h    = (float*)walloc((size_t)NNODES * HID * 4);
  float* m    = (float*)walloc((size_t)NNODES * HID * 4);
  float* agg  = (float*)walloc((size_t)NNODES * HID * 4);
  float* dinv = (float*)walloc((size_t)NNODES * 4);
  float* f    = (float*)walloc((size_t)NNODES * NFILT * 4);
  int*   dk   = (int*)  walloc((size_t)NNODES * NFILT * 4);
  float* prs  = (float*)walloc((size_t)NNODES * 32 * 4);
  __bf16* Whi[3]; __bf16* Wlo[3];
  for (int l = 0; l < 3; ++l) {
    Whi[l] = (__bf16*)walloc((size_t)HID * HID * 2);
    Wlo[l] = (__bf16*)walloc((size_t)HID * HID * 2);
  }
  __bf16* WFhi = (__bf16*)walloc((size_t)HID * NFILT * 2);
  __bf16* WFlo = (__bf16*)walloc((size_t)HID * NFILT * 2);
  __bf16* WPhi = (__bf16*)walloc((size_t)2 * NFILT * HID * 2);
  __bf16* WPlo = (__bf16*)walloc((size_t)2 * NFILT * HID * 2);

  const int T = 256;
  // pack weights -> WMMA B-fragment layout (bf16 hi/lo)
  for (int l = 0; l < 3; ++l)
    pack_b_kernel<<<(8 * 4 * 32 + T - 1) / T, T, 0, stream>>>(
        gcn_w + (size_t)l * HID * HID, Whi[l], Wlo[l], 4, 8, HID);
  pack_b_kernel<<<(1 * 4 * 32 + T - 1) / T, T, 0, stream>>>(togl_wf, WFhi, WFlo, 4, 1, NFILT);
  pack_b_kernel<<<(8 * 1 * 32 + T - 1) / T, T, 0, stream>>>(togl_wp, WPhi, WPlo, 1, 8, HID);

  // embedding
  embed_kernel<<<NNODES * 32 / T, T, 0, stream>>>(x, emb_c, emb_n, h);

  // degrees (deg = 1 + indeg), dinv = rsqrt(deg) -- recomputed each call (no state)
  fill1_kernel<<<(NNODES + T - 1) / T, T, 0, stream>>>(dinv, NNODES);
  deg_count_kernel<<<(NEDGES + T - 1) / T, T, 0, stream>>>(ei, dinv);
  rsqrt_kernel<<<(NNODES + T - 1) / T, T, 0, stream>>>(dinv, NNODES);

  const int GEMM_B = NNODES / 64;  // 4 waves * 16 rows per block
  auto gcn_layer = [&](int l, int lnIdx) {
    wmma_gemm_kernel<8><<<GEMM_B, 128, 0, stream>>>(h, Whi[l], Wlo[l], nullptr, m, 4);
    selfterm_kernel<<<NNODES * 32 / T, T, 0, stream>>>(m, dinv, agg);
    scatter_kernel<<<NEDGES * 32 / T, T, 0, stream>>>(ei, m, dinv, agg);
    ln_kernel<<<NNODES * 32 / T, T, 0, stream>>>(
        agg, gcn_b + (size_t)l * HID, ln_g + (size_t)lnIdx * HID,
        ln_b + (size_t)lnIdx * HID, h);
  };

  gcn_layer(0, 0);

  // TOGL: f = h@wf + bf ; death = scatter-max ; h += relu([f,death]@wp + bp)
  wmma_gemm_kernel<1><<<GEMM_B, 128, 0, stream>>>(h, WFhi, WFlo, togl_bf, f, 4);
  dkey_init_kernel<<<(NNODES * NFILT + T - 1) / T, T, 0, stream>>>(f, dk);
  edge_max_kernel<<<(NEDGES * NFILT + T - 1) / T, T, 0, stream>>>(ei, f, dk);
  pairs_kernel<<<(NNODES * NFILT + T - 1) / T, T, 0, stream>>>(f, dk, prs);
  wmma_gemm_kernel<8><<<GEMM_B, 128, 0, stream>>>(prs, WPhi, WPlo, togl_bp, m, 1);
  add_relu_kernel<<<(NNODES * HID / 4 + T - 1) / T, T, 0, stream>>>(m, h);

  gcn_layer(1, 0);   // reference reuses ln[0] for second layer
  gcn_layer(2, 1);

  out_kernel<<<(NGRAPH * OUTCH + T - 1) / T, T, 0, stream>>>(h, w_out, out);
}
